// Convolution_57466662420873
// MI455X (gfx1250) — matrix operand order repaired and checked by reading, not compiled, not run on hardware
//
#include <hip/hip_runtime.h>

#define GRIDSZ 128
#define CIN 32
#define COUT 64
#define TABLE_SIZE (GRIDSZ * GRIDSZ * GRIDSZ)
#define TABLE_BYTES ((size_t)TABLE_SIZE * 4)
#define NTAP 27
#define SWT 40                      // padded K-stride (bf16 elems): 80B rows -> 16B aligned, conflict-free
#define WSLAB (NTAP * COUT * SWT)   // bf16 elems in packed weight image (69120 -> 135 KB)
#define WCHUNKS (WSLAB * 2 / 16)    // 16-byte chunks (8640)
#define WAVES 8                     // 16-row tiles per 256-thread block

typedef __attribute__((ext_vector_type(16))) __bf16 v16bf;
typedef __attribute__((ext_vector_type(8)))  float  v8f;
typedef __attribute__((ext_vector_type(4)))  float  v4f;

// Types matching the async-to-LDS builtin's parameters (from hipcc diagnostic):
// arg0: 'int __vector(4) __device__ *'  -> AS(1) v4i*, non-const
// arg1: LDS destination                  -> AS(3) v4i*
typedef __attribute__((__vector_size__(16))) int v4i;
typedef __attribute__((address_space(1))) v4i* gv4i_p;
typedef __attribute__((address_space(3))) v4i* lv4i_p;

#define HAS_ASYNC_LDS __has_builtin(__builtin_amdgcn_global_load_async_to_lds_b128)

__global__ void zero_table_kernel(int* __restrict__ table) {
    for (int i = blockIdx.x * blockDim.x + threadIdx.x; i < TABLE_SIZE;
         i += gridDim.x * blockDim.x)
        table[i] = 0;
}

__global__ void build_table_kernel(const int* __restrict__ in_pos, int n_in,
                                   int* __restrict__ table) {
    int i = blockIdx.x * blockDim.x + threadIdx.x;
    if (i >= n_in) return;
    int x = in_pos[3 * i + 0];
    int y = in_pos[3 * i + 1];
    int z = in_pos[3 * i + 2];
    table[(x * GRIDSZ + y) * GRIDSZ + z] = i + 1;   // keys unique: no race
}

// Pre-pack weights into the exact LDS image: [tap][col][k] bf16, K padded to SWT.
__global__ void pack_weights_kernel(const float* __restrict__ w,   // [27][CIN][COUT]
                                    __bf16* __restrict__ wbf) {    // [27][COUT][SWT]
    int i = blockIdx.x * blockDim.x + threadIdx.x;
    if (i >= WSLAB) return;
    int k  = i % SWT;
    int c  = (i / SWT) % COUT;
    int oi = i / (SWT * COUT);
    float v = (k < CIN) ? w[((size_t)oi * CIN + k) * COUT + c] : 0.0f;
    wbf[i] = (__bf16)v;
}

__launch_bounds__(32 * WAVES)
__global__ void sparse_conv_wmma_kernel(const float*  __restrict__ features,
                                        const int*    __restrict__ out_pos,
                                        const __bf16* __restrict__ wbf,   // packed LDS image
                                        const int*    __restrict__ table,
                                        float*        __restrict__ out,
                                        int n_out) {
    __shared__ uint4 wlds4[WCHUNKS];                 // 135 KB: all 27 taps
    const __bf16* wlds = (const __bf16*)wlds4;

    // ---- one-time staging of the full weight image (async copy to LDS) ----
    const uint4* wg = (const uint4*)wbf;
#if HAS_ASYNC_LDS
    for (int i = threadIdx.x; i < WCHUNKS; i += blockDim.x)
        __builtin_amdgcn_global_load_async_to_lds_b128(
            (gv4i_p)(wg + i), (lv4i_p)(wlds4 + i), /*offset=*/0, /*cpol=*/0);
#if __has_builtin(__builtin_amdgcn_s_wait_asynccnt)
    __builtin_amdgcn_s_wait_asynccnt(0);
#else
    asm volatile("s_wait_asynccnt 0x0" ::: "memory");
#endif
#else
    for (int i = threadIdx.x; i < WCHUNKS; i += blockDim.x)
        wlds4[i] = wg[i];
#endif
    __syncthreads();                                 // only barrier in the kernel

    const int lane = threadIdx.x & 31;
    const int wave = threadIdx.x >> 5;
    const int h    = lane >> 4;       // half-wave
    const int l16  = lane & 15;
    const int tile = blockIdx.x * WAVES + wave;      // 16 output rows per tile

    int  row   = tile * 16 + l16;     // both half-waves mirror rows 0..15
    bool rowOk = row < n_out;
    int  rowc  = rowOk ? row : (n_out - 1);
    int  px = out_pos[3 * rowc + 0];
    int  py = out_pos[3 * rowc + 1];
    int  pz = out_pos[3 * rowc + 2];

    v8f acc[4];
#pragma unroll
    for (int t = 0; t < 4; ++t)
#pragma unroll
        for (int e = 0; e < 8; ++e) acc[t][e] = 0.0f;

    // 16-bit A layout (16x32): lanes 0-15 hold K 0-7 & 16-23 (M=lane),
    // lanes 16-31 hold K 8-15 & 24-31 (M=lane-16).
    const int kbaseA = h ? 8 : 0;
    // 16-bit B layout (32x16): lanes 0-15 hold K 0-7 & 16-23 (N=lane),
    // lanes 16-31 hold K 8-15 & 24-31 (N=lane-16).
    const int kbaseB = h * 8;

    for (int oi = 0; oi < NTAP; ++oi) {
        int dx = oi / 9 - 1, dy = (oi / 3) % 3 - 1, dz = oi % 3 - 1;
        int nx = px + dx, ny = py + dy, nz = pz + dz;
        bool valid = rowOk && nx >= 0 && nx < GRIDSZ && ny >= 0 && ny < GRIDSZ &&
                     nz >= 0 && nz < GRIDSZ;
        int fidx = 0;
        if (valid) fidx = table[(nx * GRIDSZ + ny) * GRIDSZ + nz];

        v16bf avec;
        if (fidx > 0) {
            const float* fp = features + (size_t)(fidx - 1) * CIN + kbaseA;
            v4f a0 = *(const v4f*)(fp + 0);    // K kbaseA+0 .. +3
            v4f a1 = *(const v4f*)(fp + 4);    // K kbaseA+4 .. +7
            v4f a2 = *(const v4f*)(fp + 16);   // K kbaseA+16.. +19
            v4f a3 = *(const v4f*)(fp + 20);   // K kbaseA+20.. +23
#pragma unroll
            for (int e = 0; e < 4; ++e) {
                avec[e + 0]  = (__bf16)a0[e];
                avec[e + 4]  = (__bf16)a1[e];
                avec[e + 8]  = (__bf16)a2[e];
                avec[e + 12] = (__bf16)a3[e];
            }
        } else {
#pragma unroll
            for (int e = 0; e < 16; ++e) avec[e] = (__bf16)0.0f;
        }

#pragma unroll
        for (int t = 0; t < 4; ++t) {
            // 16B-aligned row start: (oi*COUT+col)*SWT*2 is a multiple of 16.
            union { uint4 q[2]; v16bf v; } bu;
            const uint4* bq =
                (const uint4*)&wlds[((oi * COUT + t * 16 + l16) * SWT) + kbaseB];
            bu.q[0] = bq[0];
            bu.q[1] = bq[2];
            acc[t] = __builtin_amdgcn_wmma_f32_16x16x32_bf16(
                /*neg_a=*/false, avec, /*neg_b=*/false, bu.v,
                /*c_mod=*/(short)0, acc[t], /*reuse_a=*/false, /*reuse_b=*/false);
        }
    }

    // ---- store C/D: VGPR j -> row M = j + 8*h, col = l16 within col tile ----
#pragma unroll
    for (int t = 0; t < 4; ++t) {
#pragma unroll
        for (int j = 0; j < 8; ++j) {
            int r = tile * 16 + j + 8 * h;
            if (r < n_out) out[(size_t)r * COUT + t * 16 + l16] = acc[t][j];
        }
    }
}

extern "C" void kernel_launch(void* const* d_in, const int* in_sizes, int n_in_arrays,
                              void* d_out, int out_size, void* d_ws, size_t ws_size,
                              hipStream_t stream) {
    (void)n_in_arrays; (void)out_size; (void)ws_size;
    const float* features = (const float*)d_in[0];
    const int*   in_pos   = (const int*)  d_in[1];
    const int*   out_pos  = (const int*)  d_in[2];
    const float* weight   = (const float*)d_in[3];
    float*       out      = (float*)d_out;

    int n_in  = in_sizes[1] / 3;
    int n_out = in_sizes[2] / 3;

    int*    table = (int*)d_ws;                           // 8 MB
    __bf16* wbf   = (__bf16*)((char*)d_ws + TABLE_BYTES); // +135 KB packed weights

    zero_table_kernel<<<2048, 256, 0, stream>>>(table);
    build_table_kernel<<<(n_in + 255) / 256, 256, 0, stream>>>(in_pos, n_in, table);
    pack_weights_kernel<<<(WSLAB + 255) / 256, 256, 0, stream>>>(weight, wbf);

    int tiles  = (n_out + 15) / 16;
    int blocks = (tiles + WAVES - 1) / WAVES;
    sparse_conv_wmma_kernel<<<blocks, 32 * WAVES, 0, stream>>>(
        features, out_pos, wbf, table, out, n_out);
}
